// DANetHead_52055003628037
// MI455X (gfx1250) — matrix operand order repaired and checked by reading, not compiled, not run on hardware
//
#include <hip/hip_runtime.h>

// ---------------------------------------------------------------------------
// DANetHead forward for MI455X (gfx1250).
// bf16 WMMA for all matmuls; B-tiles streamed to LDS with
// global_load_async_to_lds_b128 (ASYNCcnt), double-buffered so the DMA
// overlaps the v_wmma stream; one barrier per K-step.
// ---------------------------------------------------------------------------

typedef __attribute__((ext_vector_type(16))) __bf16 v16bf;
typedef __attribute__((ext_vector_type(8)))  float  v8f;
typedef unsigned short u16;
typedef unsigned int   u32;

#define TM  64
#define TN  128
#define TK  32
#define LDA 40   // LDS row stride (u16): mult of 8 -> b128-aligned rows
#define LDB 40

__device__ __forceinline__ u16 f2bf(float f) {
  u32 x = __float_as_uint(f);
  u32 r = (x + 0x7FFFu + ((x >> 16) & 1u)) >> 16;  // RNE
  return (u16)r;
}

union FragBF { v16bf v; u16 u[16]; };

__device__ __forceinline__ v8f wmma_bf16(v16bf a, v16bf b, v8f c) {
  return __builtin_amdgcn_wmma_f32_16x16x32_bf16(false, a, false, b, (short)0, c,
                                                 false, false);
}

// A fragment (16x32): lane L<16 -> row M=L, K {0..7,16..23};
// lane L>=16 -> row M=L&15, K {8..15,24..31}.
__device__ __forceinline__ v16bf frag_a(const u16* sA, int mBase) {
  int lane = threadIdx.x & 31;
  const u16* p = sA + (mBase + (lane & 15)) * LDA;
  int kh = (lane >> 4) << 3;
  FragBF f;
#pragma unroll
  for (int i = 0; i < 8; ++i) f.u[i] = p[kh + i];
#pragma unroll
  for (int i = 0; i < 8; ++i) f.u[8 + i] = p[16 + kh + i];
  return f.v;
}

// B fragment (32x16) from K-contiguous LDS tile [n][k]:
// lane L<16 -> col N=L, K {0..7,16..23}; lane L>=16 -> col N=L&15, K {8..15,24..31}.
__device__ __forceinline__ v16bf frag_b(const u16* sB, int nBase) {
  int lane = threadIdx.x & 31;
  const u16* p = sB + (nBase + (lane & 15)) * LDB;
  int kh = (lane >> 4) << 3;
  FragBF f;
#pragma unroll
  for (int i = 0; i < 8; ++i) f.u[i] = p[kh + i];
#pragma unroll
  for (int i = 0; i < 8; ++i) f.u[8 + i] = p[16 + kh + i];
  return f.v;
}

// Async 16B global->LDS copy (ASYNCcnt-tracked).
__device__ __forceinline__ void async_b128(const u16* g, u16* lds) {
  unsigned l = (unsigned)(size_t)lds;  // AS3 offset = low 32 bits of flat addr
  asm volatile("global_load_async_to_lds_b128 %0, %1, off"
               :: "v"(l), "v"(g) : "memory");
}
__device__ __forceinline__ void async_wait0() {
  asm volatile("s_wait_asynccnt 0x0" ::: "memory");
}

#define ZERO4 (uint4){0u, 0u, 0u, 0u}
#define ACC_INIT(acc)                                                         \
  _Pragma("unroll") for (int i_ = 0; i_ < 2; ++i_)                            \
  _Pragma("unroll") for (int j_ = 0; j_ < 2; ++j_)                            \
      acc[i_][j_] = (v8f){0.f, 0.f, 0.f, 0.f, 0.f, 0.f, 0.f, 0.f};
#define COMPUTE_TILE(acc, sAc, sBc, wm, wn)                                   \
  {                                                                           \
    v16bf a0_ = frag_a(sAc, (wm)*32);                                         \
    v16bf a1_ = frag_a(sAc, (wm)*32 + 16);                                    \
    v16bf b0_ = frag_b(sBc, (wn)*32);                                         \
    v16bf b1_ = frag_b(sBc, (wn)*32 + 16);                                    \
    acc[0][0] = wmma_bf16(a0_, b0_, acc[0][0]);                               \
    acc[0][1] = wmma_bf16(a0_, b1_, acc[0][1]);                               \
    acc[1][0] = wmma_bf16(a1_, b0_, acc[1][0]);                               \
    acc[1][1] = wmma_bf16(a1_, b1_, acc[1][1]);                               \
  }

// ---------------------------------------------------------------------------
// Implicit-GEMM 3x3 SAME conv + BN + ReLU (double-buffered async pipeline).
// xpadT: [B][62*62][Cin] bf16 (zero halo, pixel-major).
// wtap:  [9][512][Cin] bf16.   bn: [4][512] f32.
// outF:  [B][512][3600] f32 (channel-major).
// outB:  [B][3600][512] bf16 (pixel-major).
// grid = (8, 29, B), block = 256 (8 waves).
// ---------------------------------------------------------------------------
__launch_bounds__(256)
__global__ void k_conv3_bnrelu(const u16* __restrict__ xpadT,
                               const u16* __restrict__ wtap,
                               const float* __restrict__ bn, int Cin,
                               float* __restrict__ outF, u16* __restrict__ outB) {
  __shared__ __align__(16) u16 sA[2][TM * LDA];
  __shared__ __align__(16) u16 sB[2][TN * LDB];
  __shared__ int sPix[TN];
  const int tid = threadIdx.x;
  const int mt = blockIdx.x, nt = blockIdx.y, b = blockIdx.z;
  const int Ci = 512, Npix = 3600, PP = 3844;

  if (tid < TN) {
    int n = nt * TN + tid;
    if (n < Npix) { int y = n / 60, x = n - y * 60; sPix[tid] = y * 62 + x; }
    else sPix[tid] = -1;
  }
  __syncthreads();

  // hoisted per-thread staging state (loop-invariant)
  const int akc = tid & 3, aml = tid >> 2;
  const int bkc = tid & 3, bnl0 = tid >> 2, bnl1 = 64 + bnl0;
  const int pix0 = sPix[bnl0], pix1 = sPix[bnl1];
  const int bo0 = bnl0 * LDB + bkc * 8, bo1 = bnl1 * LDB + bkc * 8;
  if (pix0 < 0) { *(uint4*)&sB[0][bo0] = ZERO4; *(uint4*)&sB[1][bo0] = ZERO4; }
  if (pix1 < 0) { *(uint4*)&sB[0][bo1] = ZERO4; *(uint4*)&sB[1][bo1] = ZERO4; }

  const u16* xb  = xpadT + (size_t)b * PP * Cin + bkc * 8;
  const u16* wa  = wtap + (size_t)(mt * TM + aml) * Cin + akc * 8;
  const size_t wtapStride = (size_t)Ci * Cin;
  const int kpc = Cin >> 5;        // K-steps per tap
  const int ksteps = 9 * kpc;

  auto stage = [&](int s, int buf) {
    int t = s / kpc;
    int k0 = (s - t * kpc) << 5;
    int off = (t / 3) * 62 + (t % 3);
    *(uint4*)&sA[buf][aml * LDA + akc * 8] =
        *(const uint4*)(wa + (size_t)t * wtapStride + k0);
    if (pix0 >= 0) async_b128(xb + (size_t)(pix0 + off) * Cin + k0, &sB[buf][bo0]);
    if (pix1 >= 0) async_b128(xb + (size_t)(pix1 + off) * Cin + k0, &sB[buf][bo1]);
  };

  const int wave = tid >> 5;
  const int wm = wave & 1, wn = wave >> 1;
  v8f acc[2][2];
  ACC_INIT(acc);

  stage(0, 0);
  async_wait0();
  __syncthreads();
  for (int s = 0; s < ksteps; ++s) {
    int cur = s & 1;
    if (s + 1 < ksteps) stage(s + 1, cur ^ 1);   // DMA overlaps WMMA below
    COMPUTE_TILE(acc, &sA[cur][0], &sB[cur][0], wm, wn);
    async_wait0();
    __syncthreads();
  }

  const int lane = tid & 31;
  const int nOff = lane & 15, mOff = (lane >> 4) * 8;
#pragma unroll
  for (int fi = 0; fi < 2; ++fi)
#pragma unroll
    for (int fj = 0; fj < 2; ++fj) {
      int mBase = mt * TM + wm * 32 + fi * 16 + mOff;
      int nIdx  = nt * TN + wn * 32 + fj * 16 + nOff;
      if (nIdx >= Npix) continue;
#pragma unroll
      for (int r = 0; r < 8; ++r) {
        int m = mBase + r;
        float s  = bn[m],        bb = bn[512 + m];
        float mu = bn[1024 + m], vr = bn[1536 + m];
        float y = s * (acc[fi][fj][r] - mu) * rsqrtf(vr + 1e-5f) + bb;
        y = fmaxf(y, 0.f);
        outF[(size_t)b * Ci * Npix + (size_t)m * Npix + nIdx] = y;
        outB[((size_t)b * Npix + nIdx) * Ci + m] = f2bf(y);
      }
    }
}

// ---------------------------------------------------------------------------
// Generic GEMM: C[M,N] = A[M,K] * Bt[N,K]^T  (B rows K-contiguous).
// mode 0: outB[b][n][m] = bf16(acc + bias[m])   (pixel-major, q/k)
// mode 1: outB[b][m][n] = bf16(acc + bias[m])   (channel-major, v)
// mode 2: outF[b][m][n] = acc + bias[m]         (output heads)
// mode 3: outF[b][m][n] = scale[0]*acc + resid[b][m][n]   (AV / channel-apply)
// ---------------------------------------------------------------------------
__launch_bounds__(256)
__global__ void k_gemm_bt(const u16* __restrict__ A, const u16* __restrict__ Bt,
                          int M, int N, int K,
                          long sAb, long sBb, long sOb, long sRb,
                          const float* __restrict__ bias,
                          const float* __restrict__ scale,
                          const float* __restrict__ resid,
                          u16* __restrict__ outB, float* __restrict__ outF,
                          int mode) {
  __shared__ __align__(16) u16 sA[2][TM * LDA];
  __shared__ __align__(16) u16 sB[2][TN * LDB];
  const int tid = threadIdx.x;
  const int mt = blockIdx.x, nt = blockIdx.y, b = blockIdx.z;
  const u16* Ab = A  + (size_t)b * sAb;
  const u16* Bb = Bt + (size_t)b * sBb;

  const int akc = tid & 3, aml = tid >> 2;
  const int bkc = tid & 3, bnl0 = tid >> 2, bnl1 = 64 + bnl0;
  const int n0 = nt * TN + bnl0, n1 = nt * TN + bnl1;
  const bool nv0 = n0 < N, nv1 = n1 < N;
  const int bo0 = bnl0 * LDB + bkc * 8, bo1 = bnl1 * LDB + bkc * 8;
  if (!nv0) { *(uint4*)&sB[0][bo0] = ZERO4; *(uint4*)&sB[1][bo0] = ZERO4; }
  if (!nv1) { *(uint4*)&sB[0][bo1] = ZERO4; *(uint4*)&sB[1][bo1] = ZERO4; }

  const int am = mt * TM + aml;
  const bool av = am < M;
  const u16* A0 = Ab + (size_t)am * K + akc * 8;
  const u16* B0 = Bb + (size_t)n0 * K + bkc * 8;
  const u16* B1 = Bb + (size_t)n1 * K + bkc * 8;

  auto stage = [&](int k0, int buf) {
    uint4 v = ZERO4;
    if (av && (k0 + akc * 8) < K) v = *(const uint4*)(A0 + k0);
    *(uint4*)&sA[buf][aml * LDA + akc * 8] = v;
    bool kv = (k0 + bkc * 8) < K;   // K mult of 8 -> chunk-granular
    if (nv0) { if (kv) async_b128(B0 + k0, &sB[buf][bo0]);
               else    *(uint4*)&sB[buf][bo0] = ZERO4; }
    if (nv1) { if (kv) async_b128(B1 + k0, &sB[buf][bo1]);
               else    *(uint4*)&sB[buf][bo1] = ZERO4; }
  };

  const int wave = tid >> 5;
  const int wm = wave & 1, wn = wave >> 1;
  v8f acc[2][2];
  ACC_INIT(acc);

  const int nsteps = (K + TK - 1) / TK;
  stage(0, 0);
  async_wait0();
  __syncthreads();
  for (int s = 0; s < nsteps; ++s) {
    int cur = s & 1;
    if (s + 1 < nsteps) stage((s + 1) * TK, cur ^ 1);
    COMPUTE_TILE(acc, &sA[cur][0], &sB[cur][0], wm, wn);
    async_wait0();
    __syncthreads();
  }

  const int lane = tid & 31;
  const int nOff = lane & 15, mOff = (lane >> 4) * 8;
  float scv = (mode == 3) ? scale[0] : 0.f;
#pragma unroll
  for (int fi = 0; fi < 2; ++fi)
#pragma unroll
    for (int fj = 0; fj < 2; ++fj) {
      int mBase = mt * TM + wm * 32 + fi * 16 + mOff;
      int nIdx  = nt * TN + wn * 32 + fj * 16 + nOff;
      if (nIdx >= N) continue;
#pragma unroll
      for (int r = 0; r < 8; ++r) {
        int m = mBase + r;
        if (m >= M) continue;
        float a = acc[fi][fj][r];
        if (mode == 0)
          outB[(size_t)b * sOb + (size_t)nIdx * M + m] = f2bf(a + bias[m]);
        else if (mode == 1)
          outB[(size_t)b * sOb + (size_t)m * N + nIdx] = f2bf(a + bias[m]);
        else if (mode == 2)
          outF[(size_t)b * sOb + (size_t)m * N + nIdx] = a + bias[m];
        else
          outF[(size_t)b * sOb + (size_t)m * N + nIdx] =
              scv * a + resid[(size_t)b * sRb + (size_t)m * N + nIdx];
      }
    }
}

// ---------------------------------------------------------------------------
// energy[n,m] = sum_c q[n][c]*k[m][c] + lamb*(d[n]-d[m])^2
// q,k pixel-major [3600][64] -> A row-major, B K-contiguous.  K=64.
// ---------------------------------------------------------------------------
__launch_bounds__(256)
__global__ void k_energy(const u16* __restrict__ q, const u16* __restrict__ kk,
                         const float* __restrict__ dmap,
                         const float* __restrict__ plamb,
                         float* __restrict__ outE) {
  __shared__ __align__(16) u16 sA[2][TM * LDA];
  __shared__ __align__(16) u16 sB[2][TN * LDB];
  const int Npix = 3600, Cq = 64;
  const int tid = threadIdx.x;
  const int mt = blockIdx.x, nt = blockIdx.y, b = blockIdx.z;
  const u16* qb = q  + (size_t)b * Npix * Cq;
  const u16* kb = kk + (size_t)b * Npix * Cq;

  const int akc = tid & 3, aml = tid >> 2;
  const int bkc = tid & 3, bnl0 = tid >> 2, bnl1 = 64 + bnl0;
  const int n0 = nt * TN + bnl0, n1 = nt * TN + bnl1;
  const bool nv0 = n0 < Npix, nv1 = n1 < Npix;
  const int bo0 = bnl0 * LDB + bkc * 8, bo1 = bnl1 * LDB + bkc * 8;
  if (!nv0) { *(uint4*)&sB[0][bo0] = ZERO4; *(uint4*)&sB[1][bo0] = ZERO4; }
  if (!nv1) { *(uint4*)&sB[0][bo1] = ZERO4; *(uint4*)&sB[1][bo1] = ZERO4; }

  const int am = mt * TM + aml;
  const bool av = am < Npix;
  const u16* A0 = qb + (size_t)am * Cq + akc * 8;
  const u16* B0 = kb + (size_t)n0 * Cq + bkc * 8;
  const u16* B1 = kb + (size_t)n1 * Cq + bkc * 8;

  auto stage = [&](int k0, int buf) {
    uint4 v = ZERO4;
    if (av) v = *(const uint4*)(A0 + k0);
    *(uint4*)&sA[buf][aml * LDA + akc * 8] = v;
    if (nv0) async_b128(B0 + k0, &sB[buf][bo0]);
    if (nv1) async_b128(B1 + k0, &sB[buf][bo1]);
  };

  const int wave = tid >> 5;
  const int wm = wave & 1, wn = wave >> 1;
  v8f acc[2][2];
  ACC_INIT(acc);

  stage(0, 0);
  async_wait0();
  __syncthreads();
#pragma unroll
  for (int s = 0; s < 2; ++s) {   // Cq/TK == 2
    int cur = s & 1;
    if (s + 1 < 2) stage((s + 1) * TK, cur ^ 1);
    COMPUTE_TILE(acc, &sA[cur][0], &sB[cur][0], wm, wn);
    async_wait0();
    __syncthreads();
  }

  const int lane = tid & 31;
  const int nOff = lane & 15, mOff = (lane >> 4) * 8;
  const float lamb = plamb[0];
  const float* db = dmap + (size_t)b * Npix;
#pragma unroll
  for (int fi = 0; fi < 2; ++fi)
#pragma unroll
    for (int fj = 0; fj < 2; ++fj) {
      int mBase = mt * TM + wm * 32 + fi * 16 + mOff;
      int nIdx  = nt * TN + wn * 32 + fj * 16 + nOff;
      if (nIdx >= Npix) continue;
      float dn = db[nIdx];
#pragma unroll
      for (int r = 0; r < 8; ++r) {
        int m = mBase + r;
        if (m >= Npix) continue;
        float dd = db[m] - dn;
        outE[(size_t)b * Npix * Npix + (size_t)m * Npix + nIdx] =
            acc[fi][fj][r] + lamb * dd * dd;
      }
    }
}

// ---------------------------------------------------------------------------
// Gram matrix: cen[c,d] = sum_n F[n][c]*F[n][d];  F pixel-major [3600][512].
// Both operands K-strided -> scalar (thread-coalesced) staging.
// ---------------------------------------------------------------------------
__launch_bounds__(256)
__global__ void k_gram(const u16* __restrict__ F, float* __restrict__ outF) {
  __shared__ __align__(16) u16 sA[TM * LDA];
  __shared__ __align__(16) u16 sB[TN * LDB];
  const int Ci = 512, NP = 3600;
  const int tid = threadIdx.x;
  const int mt = blockIdx.x, nt = blockIdx.y, b = blockIdx.z;
  const u16* Fb = F + (size_t)b * NP * Ci;

  const int wave = tid >> 5;
  const int wm = wave & 1, wn = wave >> 1;
  v8f acc[2][2];
  ACC_INIT(acc);

  for (int k0 = 0; k0 < NP; k0 += TK) {
#pragma unroll
    for (int i = 0; i < (TM * TK) / 256; ++i) {  // m fastest -> coalesced
      int idx = i * 256 + tid;
      int ml = idx & 63, kl = idx >> 6;
      int k = k0 + kl;
      sA[ml * LDA + kl] = (k < NP) ? Fb[(size_t)k * Ci + mt * TM + ml] : (u16)0;
    }
#pragma unroll
    for (int i = 0; i < (TN * TK) / 256; ++i) {
      int idx = i * 256 + tid;
      int nl = idx & 127, kl = idx >> 7;
      int k = k0 + kl;
      sB[nl * LDB + kl] = (k < NP) ? Fb[(size_t)k * Ci + nt * TN + nl] : (u16)0;
    }
    __syncthreads();
    COMPUTE_TILE(acc, sA, sB, wm, wn);
    __syncthreads();
  }

  const int lane = tid & 31;
  const int nOff = lane & 15, mOff = (lane >> 4) * 8;
#pragma unroll
  for (int fi = 0; fi < 2; ++fi)
#pragma unroll
    for (int fj = 0; fj < 2; ++fj) {
      int mBase = mt * TM + wm * 32 + fi * 16 + mOff;
      int nIdx  = nt * TN + wn * 32 + fj * 16 + nOff;
#pragma unroll
      for (int r = 0; r < 8; ++r) {
        int m = mBase + r;
        outF[(size_t)b * Ci * Ci + (size_t)m * Ci + nIdx] = acc[fi][fj][r];
      }
    }
}

// ---------------------------------------------------------------------------
// Softmax helpers
// ---------------------------------------------------------------------------
__launch_bounds__(256)
__global__ void k_softmax_row(const float* __restrict__ E, u16* __restrict__ att,
                              int L) {
  const int row = blockIdx.x, tid = threadIdx.x;
  const float* e = E + (size_t)row * L;
  __shared__ float red[256];
  float mx = -3.4e38f;
  for (int i = tid; i < L; i += 256) mx = fmaxf(mx, e[i]);
  red[tid] = mx; __syncthreads();
  for (int s = 128; s > 0; s >>= 1) {
    if (tid < s) red[tid] = fmaxf(red[tid], red[tid + s]);
    __syncthreads();
  }
  mx = red[0]; __syncthreads();
  float sum = 0.f;
  for (int i = tid; i < L; i += 256) sum += __expf(e[i] - mx);
  red[tid] = sum; __syncthreads();
  for (int s = 128; s > 0; s >>= 1) {
    if (tid < s) red[tid] += red[tid + s];
    __syncthreads();
  }
  float inv = 1.f / red[0];
  u16* o = att + (size_t)row * L;
  for (int i = tid; i < L; i += 256) o[i] = f2bf(__expf(e[i] - mx) * inv);
}

// softmax(rowmax - cen) == exp(rowmin - cen)/sum
__launch_bounds__(256)
__global__ void k_chan_softmax(const float* __restrict__ cen,
                               u16* __restrict__ catt, int L) {
  const int row = blockIdx.x, tid = threadIdx.x;
  const float* e = cen + (size_t)row * L;
  __shared__ float red[256];
  float mn = 3.4e38f;
  for (int i = tid; i < L; i += 256) mn = fminf(mn, e[i]);
  red[tid] = mn; __syncthreads();
  for (int s = 128; s > 0; s >>= 1) {
    if (tid < s) red[tid] = fminf(red[tid], red[tid + s]);
    __syncthreads();
  }
  mn = red[0]; __syncthreads();
  float sum = 0.f;
  for (int i = tid; i < L; i += 256) sum += __expf(mn - e[i]);
  red[tid] = sum; __syncthreads();
  for (int s = 128; s > 0; s >>= 1) {
    if (tid < s) red[tid] += red[tid + s];
    __syncthreads();
  }
  float inv = 1.f / red[0];
  u16* o = catt + (size_t)row * L;
  for (int i = tid; i < L; i += 256) o[i] = f2bf(__expf(mn - e[i]) * inv);
}

// ---------------------------------------------------------------------------
// Elementwise helpers
// ---------------------------------------------------------------------------
// src: f32 channel-major [B*Cch][3600] -> dst: bf16 pixel-major [B][3844][Cch]
__global__ void k_pad_t(const float* __restrict__ src, u16* __restrict__ dst,
                        int Bn, int Cch) {
  int idx = blockIdx.x * 256 + threadIdx.x;
  if (idx >= Bn * 3844 * Cch) return;
  int c = idx % Cch, t = idx / Cch;
  int p = t % 3844, b = t / 3844;
  int r = p / 62, col = p - r * 62;
  u16 v = 0;
  if (r >= 1 && r <= 60 && col >= 1 && col <= 60)
    v = f2bf(src[((size_t)b * Cch + c) * 3600 + (r - 1) * 60 + (col - 1)]);
  dst[idx] = v;
}

__global__ void k_cvt(const float* __restrict__ s, u16* __restrict__ d, int n) {
  int i = blockIdx.x * 256 + threadIdx.x;
  if (i < n) d[i] = f2bf(s[i]);
}

// w [M][K][3][3] f32 -> wt [9][M][K] bf16
__global__ void k_wtap(const float* __restrict__ w, u16* __restrict__ wt,
                       int M, int K) {
  int idx = blockIdx.x * 256 + threadIdx.x;
  if (idx >= M * K * 9) return;
  int t = idx % 9, r = idx / 9;
  wt[(size_t)t * M * K + r] = f2bf(w[(size_t)r * 9 + t]);
}

// channel-major f32 a+b -> pixel-major bf16
__global__ void k_add_t(const float* __restrict__ a, const float* __restrict__ b,
                        u16* __restrict__ d, int Bn, int C, int N) {
  int idx = blockIdx.x * 256 + threadIdx.x;
  if (idx >= Bn * C * N) return;
  int c = idx % C, t = idx / C;
  int n = t % N, bb = t / N;
  size_t src = ((size_t)bb * C + c) * N + n;
  d[idx] = f2bf(a[src] + b[src]);
}

// ---------------------------------------------------------------------------
// Host orchestration
// ---------------------------------------------------------------------------
extern "C" void kernel_launch(void* const* d_in, const int* in_sizes, int n_in,
                              void* d_out, int out_size, void* d_ws,
                              size_t ws_size, hipStream_t stream) {
  (void)in_sizes; (void)n_in; (void)out_size; (void)ws_size;
  const float* x     = (const float*)d_in[0];
  const float* dmap  = (const float*)d_in[1];
  const float* w_s0  = (const float*)d_in[2];
  const float* bn_s0 = (const float*)d_in[3];
  const float* wq    = (const float*)d_in[4];
  const float* bq    = (const float*)d_in[5];
  const float* wk    = (const float*)d_in[6];
  const float* bk    = (const float*)d_in[7];
  const float* wv    = (const float*)d_in[8];
  const float* bv    = (const float*)d_in[9];
  const float* gsa   = (const float*)d_in[10];
  const float* lamb  = (const float*)d_in[11];
  const float* w_s1  = (const float*)d_in[12];
  const float* bn_s1 = (const float*)d_in[13];
  const float* w6    = (const float*)d_in[14];
  const float* b6    = (const float*)d_in[15];
  const float* w_c0  = (const float*)d_in[16];
  const float* bn_c0 = (const float*)d_in[17];
  const float* gsc   = (const float*)d_in[18];
  const float* w_c1  = (const float*)d_in[19];
  const float* bn_c1 = (const float*)d_in[20];
  const float* w7    = (const float*)d_in[21];
  const float* b7    = (const float*)d_in[22];
  const float* w8    = (const float*)d_in[23];
  const float* b8    = (const float*)d_in[24];
  float* out = (float*)d_out;

  const int B = 2, Cin = 2048, Ci = 512, Cq = 64, Co = 40, NP = 3600, PP = 3844;

  char* ws = (char*)d_ws;
  size_t off = 0;
  auto alloc = [&](size_t bytes) -> void* {
    off = (off + 255) & ~(size_t)255;
    void* p = ws + off;
    off += bytes;
    return p;
  };

  u16* xpadT = (u16*)alloc((size_t)B * PP * Cin * 2);   // pixel-major padded x
  u16* wt_s0 = (u16*)alloc((size_t)9 * Ci * Cin * 2);
  u16* wt_c0 = (u16*)alloc((size_t)9 * Ci * Cin * 2);
  u16* wt_s1 = (u16*)alloc((size_t)9 * Ci * Ci * 2);
  u16* wt_c1 = (u16*)alloc((size_t)9 * Ci * Ci * 2);
  u16* wq_b  = (u16*)alloc((size_t)Cq * Ci * 2);
  u16* wk_b  = (u16*)alloc((size_t)Cq * Ci * 2);
  u16* wv_b  = (u16*)alloc((size_t)Ci * Ci * 2);
  u16* w6_b  = (u16*)alloc((size_t)Co * Ci * 2);
  u16* w7_b  = (u16*)alloc((size_t)Co * Ci * 2);
  u16* w8_b  = (u16*)alloc((size_t)Co * Ci * 2);
  float* f1_f = (float*)alloc((size_t)B * Ci * NP * 4);  // channel-major
  u16*   f1_b = (u16*)alloc((size_t)B * NP * Ci * 2);    // pixel-major
  float* f2_f = (float*)alloc((size_t)B * Ci * NP * 4);
  u16*   f2_b = (u16*)alloc((size_t)B * NP * Ci * 2);
  u16* q_b = (u16*)alloc((size_t)B * NP * Cq * 2);       // pixel-major
  u16* k_b = (u16*)alloc((size_t)B * NP * Cq * 2);       // pixel-major
  u16* v_b = (u16*)alloc((size_t)B * Ci * NP * 2);       // channel-major
  float* sa_f  = (float*)alloc((size_t)B * Ci * NP * 4);
  u16*   sa_p  = (u16*)alloc((size_t)B * PP * Ci * 2);
  float* sac_f = (float*)alloc((size_t)B * Ci * NP * 4);
  u16*   sac_b = (u16*)alloc((size_t)B * NP * Ci * 2);
  u16*   fsum_b = (u16*)alloc((size_t)B * NP * Ci * 2);
  u16*   att    = (u16*)alloc((size_t)B * NP * NP * 2);
  float* energy = (float*)alloc((size_t)B * NP * NP * 4);
  // channel-branch scratch aliases the (dead-after-softmax) energy buffer
  char* ali = (char*)energy;
  size_t ao = 0;
  auto aalloc = [&](size_t bytes) -> void* {
    ao = (ao + 255) & ~(size_t)255;
    void* p = ali + ao;
    ao += bytes;
    return p;
  };
  float* cen   = (float*)aalloc((size_t)B * Ci * Ci * 4);
  u16*   catt  = (u16*)aalloc((size_t)B * Ci * Ci * 2);
  float* sc_f  = (float*)aalloc((size_t)B * Ci * NP * 4);
  u16*   sc_p  = (u16*)aalloc((size_t)B * PP * Ci * 2);
  float* scc_f = (float*)aalloc((size_t)B * Ci * NP * 4);
  u16*   scc_b = (u16*)aalloc((size_t)B * NP * Ci * 2);

  // ---- prep -------------------------------------------------------------
  {
    int t = 9 * Ci * Cin;
    k_wtap<<<(t + 255) / 256, 256, 0, stream>>>(w_s0, wt_s0, Ci, Cin);
    k_wtap<<<(t + 255) / 256, 256, 0, stream>>>(w_c0, wt_c0, Ci, Cin);
    t = 9 * Ci * Ci;
    k_wtap<<<(t + 255) / 256, 256, 0, stream>>>(w_s1, wt_s1, Ci, Ci);
    k_wtap<<<(t + 255) / 256, 256, 0, stream>>>(w_c1, wt_c1, Ci, Ci);
  }
  k_cvt<<<(Cq * Ci + 255) / 256, 256, 0, stream>>>(wq, wq_b, Cq * Ci);
  k_cvt<<<(Cq * Ci + 255) / 256, 256, 0, stream>>>(wk, wk_b, Cq * Ci);
  k_cvt<<<(Ci * Ci + 255) / 256, 256, 0, stream>>>(wv, wv_b, Ci * Ci);
  k_cvt<<<(Co * Ci + 255) / 256, 256, 0, stream>>>(w6, w6_b, Co * Ci);
  k_cvt<<<(Co * Ci + 255) / 256, 256, 0, stream>>>(w7, w7_b, Co * Ci);
  k_cvt<<<(Co * Ci + 255) / 256, 256, 0, stream>>>(w8, w8_b, Co * Ci);
  k_pad_t<<<((size_t)B * PP * Cin + 255) / 256, 256, 0, stream>>>(x, xpadT, B, Cin);

  const dim3 blk(256);
  const dim3 cg(Ci / TM, (NP + TN - 1) / TN, B);            // (8, 29, 2)
  const dim3 g1(1, (NP + TN - 1) / TN, B);                  // M = 64 / 40
  const dim3 ge((NP + TM - 1) / TM, (NP + TN - 1) / TN, B); // (57, 29, 2)
  const dim3 gg(Ci / TM, Ci / TN, B);                       // (8, 4, 2)

  // ---- position-attention branch ---------------------------------------
  k_conv3_bnrelu<<<cg, blk, 0, stream>>>(xpadT, wt_s0, bn_s0, Cin, f1_f, f1_b);
  k_gemm_bt<<<g1, blk, 0, stream>>>(wq_b, f1_b, Cq, NP, Ci,
                                    0, (long)NP * Ci, (long)NP * Cq, 0,
                                    bq, nullptr, nullptr, q_b, nullptr, 0);
  k_gemm_bt<<<g1, blk, 0, stream>>>(wk_b, f1_b, Cq, NP, Ci,
                                    0, (long)NP * Ci, (long)NP * Cq, 0,
                                    bk, nullptr, nullptr, k_b, nullptr, 0);
  k_gemm_bt<<<cg, blk, 0, stream>>>(wv_b, f1_b, Ci, NP, Ci,
                                    0, (long)NP * Ci, (long)Ci * NP, 0,
                                    bv, nullptr, nullptr, v_b, nullptr, 1);
  k_energy<<<ge, blk, 0, stream>>>(q_b, k_b, dmap, lamb, energy);
  k_softmax_row<<<B * NP, blk, 0, stream>>>(energy, att, NP);
  k_gemm_bt<<<cg, blk, 0, stream>>>(v_b, att, Ci, NP, NP,
                                    (long)Ci * NP, (long)NP * NP,
                                    (long)Ci * NP, (long)Ci * NP,
                                    nullptr, gsa, f1_f, nullptr, sa_f, 3);
  k_pad_t<<<((size_t)B * PP * Ci + 255) / 256, 256, 0, stream>>>(sa_f, sa_p, B, Ci);
  k_conv3_bnrelu<<<cg, blk, 0, stream>>>(sa_p, wt_s1, bn_s1, Ci, sac_f, sac_b);
  k_gemm_bt<<<g1, blk, 0, stream>>>(w6_b, sac_b, Co, NP, Ci,
                                    0, (long)NP * Ci, (long)Co * NP, 0,
                                    b6, nullptr, nullptr, nullptr,
                                    out + (size_t)B * Co * NP, 2);  // sa_output

  // ---- channel-attention branch (energy region dead now) ----------------
  k_conv3_bnrelu<<<cg, blk, 0, stream>>>(xpadT, wt_c0, bn_c0, Cin, f2_f, f2_b);
  k_gram<<<gg, blk, 0, stream>>>(f2_b, cen);
  k_chan_softmax<<<B * Ci, blk, 0, stream>>>(cen, catt, Ci);
  k_gemm_bt<<<cg, blk, 0, stream>>>(catt, f2_b, Ci, NP, Ci,
                                    (long)Ci * Ci, (long)NP * Ci,
                                    (long)Ci * NP, (long)Ci * NP,
                                    nullptr, gsc, f2_f, nullptr, sc_f, 3);
  k_pad_t<<<((size_t)B * PP * Ci + 255) / 256, 256, 0, stream>>>(sc_f, sc_p, B, Ci);
  k_conv3_bnrelu<<<cg, blk, 0, stream>>>(sc_p, wt_c1, bn_c1, Ci, scc_f, scc_b);
  k_gemm_bt<<<g1, blk, 0, stream>>>(w7_b, scc_b, Co, NP, Ci,
                                    0, (long)NP * Ci, (long)Co * NP, 0,
                                    b7, nullptr, nullptr, nullptr,
                                    out + (size_t)2 * B * Co * NP, 2); // sc_out

  // ---- fusion head ------------------------------------------------------
  k_add_t<<<((size_t)B * NP * Ci + 255) / 256, 256, 0, stream>>>(
      sac_f, scc_f, fsum_b, B, Ci, NP);
  k_gemm_bt<<<g1, blk, 0, stream>>>(w8_b, fsum_b, Co, NP, Ci,
                                    0, (long)NP * Ci, (long)Co * NP, 0,
                                    b8, nullptr, nullptr, nullptr, out, 2); // sasc
}
